// Cy2MixerBlock_6700148982551
// MI455X (gfx1250) — matrix operand order repaired and checked by reading, not compiled, not run on hardware
//
#include <hip/hip_runtime.h>
#include <math.h>

// ---------------------------------------------------------------------------
// Cy2Mixer block for MI455X (gfx1250, wave32, WMMA bf16 16x16x32, f32 accum)
// K/V/vn panels staged into LDS per workgroup via the Tensor Data Mover.
// ---------------------------------------------------------------------------

typedef __bf16 bf16;
typedef unsigned int u32;
typedef __attribute__((ext_vector_type(16))) __bf16 bf16x16;
typedef __attribute__((ext_vector_type(8)))  float  f32x8;
typedef __attribute__((ext_vector_type(4)))  u32    u32x4;
typedef __attribute__((ext_vector_type(8)))  int    i32x8;
typedef __attribute__((ext_vector_type(4)))  int    i32x4;

#define B_   32
#define T_   12
#define N_   325
#define D_   128
#define F_   128
#define NP   352               // padded node count: 22*16 = 11*32
#define NBT  (B_ * T_)         // 384
#define NTOK (B_ * T_ * N_)    // 124800

#define WMMA_BF16(A, Bm, C) \
  __builtin_amdgcn_wmma_f32_16x16x32_bf16(false, (A), false, (Bm), (short)0, (C), false, false)

#if defined(__has_builtin)
#if __has_builtin(__builtin_amdgcn_tensor_load_to_lds) && __has_builtin(__builtin_amdgcn_s_wait_tensorcnt)
#define HAS_TDM 1
#endif
#endif
#ifndef HAS_TDM
#define HAS_TDM 0
#endif

__device__ __forceinline__ int lane_id() { return (int)(threadIdx.x & 31u); }

__device__ __forceinline__ f32x8 zero8() {
  f32x8 z = {0.f, 0.f, 0.f, 0.f, 0.f, 0.f, 0.f, 0.f};
  return z;
}

#if HAS_TDM
// TDM: async 2D tile (dim0 contiguous, dim1 rows) of bf16 from global -> LDS.
// Descriptor layout per CDNA5 ISA (D# groups 0/1).
__device__ __forceinline__ void tdm_load_2d_bf16(const void* gaddr, u32 lds_off,
                                                 u32 dim0, u32 dim1, u32 stride0) {
  const unsigned long long ga = (unsigned long long)gaddr;
  u32x4 g0;
  g0[0] = 1u;                                            // count=1, user descriptor
  g0[1] = lds_off;                                       // lds_addr (bytes)
  g0[2] = (u32)(ga & 0xffffffffu);                       // global_addr[31:0]
  g0[3] = (u32)((ga >> 32) & 0x01ffffffu) | (2u << 30);  // global_addr[56:32] | type=2
  i32x8 g1;
  g1[0] = (int)(1u << 16);                               // data_size = 1 -> 2 bytes
  g1[1] = (int)(dim0 << 16);                             // tensor_dim0[15:0]
  g1[2] = (int)((dim0 >> 16) | (dim1 << 16));            // tensor_dim0 hi | tensor_dim1 lo
  g1[3] = (int)((dim1 >> 16) | (dim0 << 16));            // tensor_dim1 hi | tile_dim0
  g1[4] = (int)(dim1 & 0xffffu);                         // tile_dim1 (tile_dim2 = 0)
  g1[5] = (int)stride0;                                  // tensor_dim0_stride[31:0]
  g1[6] = 0;
  g1[7] = 0;
  i32x4 z4 = {0, 0, 0, 0};
#if defined(__clang_major__) && (__clang_major__ >= 23)
  i32x8 z8 = {0, 0, 0, 0, 0, 0, 0, 0};
  __builtin_amdgcn_tensor_load_to_lds(g0, g1, z4, z4, z8, 0);
#else
  __builtin_amdgcn_tensor_load_to_lds(g0, g1, z4, z4, 0);
#endif
}
#endif

// A fragment: 16(M) x 32(K) bf16, row-major source, leading dim lda. (ISA 7.12.2)
__device__ __forceinline__ bf16x16 load_a_bf16(const bf16* A, int lda) {
  const int l = lane_id();
  const int m = l & 15;
  const int kh = l >> 4;
  bf16x16 a;
#pragma unroll
  for (int v = 0; v < 8; ++v) {
    const int k = 16 * kh + 2 * v;
    a[2 * v]     = A[m * lda + k];
    a[2 * v + 1] = A[m * lda + k + 1];
  }
  return a;
}

// B fragment: 32(K) x 16(N) bf16, row-major source, leading dim ldb.
__device__ __forceinline__ bf16x16 load_b_bf16(const bf16* Bp, int ldb) {
  const int l = lane_id();
  const int n = l & 15;
  const int kh = l >> 4;
  bf16x16 b;
#pragma unroll
  for (int v = 0; v < 8; ++v) {
    const int k = 16 * kh + 2 * v;
    b[2 * v]     = Bp[k * ldb + n];
    b[2 * v + 1] = Bp[(k + 1) * ldb + n];
  }
  return b;
}

// ---------------------------------------------------------------------------
// k0a: repack conv_w (N,N,1,3) fp32 -> 3 x (NP x NP) bf16, zero padded
// ---------------------------------------------------------------------------
__global__ __launch_bounds__(256) void k0_prep_w(const float* __restrict__ conv_w,
                                                 bf16* __restrict__ wpad) {
  long i = (long)blockIdx.x * 256 + threadIdx.x;
  const long tot = 3L * NP * NP;
  if (i >= tot) return;
  const int dk  = (int)(i / (NP * NP));
  const int rem = (int)(i % (NP * NP));
  const int n = rem / NP, m = rem % NP;
  float v = (n < N_ && m < N_) ? conv_w[((long)n * N_ + m) * 3 + dk] : 0.f;
  wpad[i] = (bf16)v;
}

// ---------------------------------------------------------------------------
// k0b: one-time fp32 -> bf16 conversion of all dense weights
// ---------------------------------------------------------------------------
__global__ __launch_bounds__(256) void k0_prep_dense(
    const float* __restrict__ Wqkv, const float* __restrict__ W1,
    const float* __restrict__ Wo, const float* __restrict__ W2,
    bf16* __restrict__ wqkvb, bf16* __restrict__ w1b,
    bf16* __restrict__ wob, bf16* __restrict__ w2b) {
  const int i = blockIdx.x * 256 + threadIdx.x;
  if (i < D_ * 384) wqkvb[i] = (bf16)Wqkv[i];
  if (i < D_ * 256) w1b[i] = (bf16)W1[i];
  if (i < D_ * 128) { wob[i] = (bf16)Wo[i]; w2b[i] = (bf16)W2[i]; }
}

// ---------------------------------------------------------------------------
// k0c: zero the pad rows [N_, NP) of q/k/v/vn workspaces
// ---------------------------------------------------------------------------
__global__ __launch_bounds__(256) void k0_zero_pad(bf16* __restrict__ qw, bf16* __restrict__ kw,
                                                   bf16* __restrict__ vw, bf16* __restrict__ vnw) {
  long i = (long)blockIdx.x * 256 + threadIdx.x;
  const long tot = (long)NBT * (NP - N_) * F_;
  if (i >= tot) return;
  const int f  = (int)(i % F_);
  const long r = i / F_;
  const int pr = (int)(r % (NP - N_));
  const int bt = (int)(r / (NP - N_));
  const long idx = ((long)bt * NP + N_ + pr) * F_ + f;
  const bf16 z = (bf16)0.0f;
  qw[idx] = z; kw[idx] = z; vw[idx] = z; vnw[idx] = z;
}

// ---------------------------------------------------------------------------
// k1: fused LayerNorm + coords + QKV proj + FC1(+coord affine, ReLU) + SGU-LN
// ---------------------------------------------------------------------------
__global__ __launch_bounds__(128) void k1_ln_proj(
    const float* __restrict__ x,
    const float* __restrict__ norm_g, const float* __restrict__ norm_b,
    const bf16* __restrict__ w1b, const float* __restrict__ b1,
    const float* __restrict__ aff1_W, const float* __restrict__ aff1_b,
    const float* __restrict__ Wl, const float* __restrict__ bl,
    const float* __restrict__ sgu_g, const float* __restrict__ sgu_b,
    const bf16* __restrict__ wqkvb, const float* __restrict__ bqkv,
    bf16* __restrict__ qw, bf16* __restrict__ kw, bf16* __restrict__ vw,
    bf16* __restrict__ vnw, float* __restrict__ uw) {
  __shared__ float xs[16][D_ + 4];
  __shared__ bf16  xb[16][D_ + 8];
  __shared__ float vvs[16][D_ + 4];
  __shared__ float co[16][2];

  const int tid = threadIdx.x;
  const long g0 = (long)blockIdx.x * 16;

  for (int i = tid; i < 16 * D_; i += 128) {
    const int r = i >> 7, c = i & 127;
    xs[r][c] = x[(g0 + r) * D_ + c];
  }
  __syncthreads();

  if (tid < 16) {
    float m = 0.f, s = 0.f;
    for (int c = 0; c < D_; ++c) { const float v = xs[tid][c]; m += v; s += v * v; }
    m *= (1.f / D_);
    s = s * (1.f / D_) - m * m;
    const float inv = rsqrtf(s + 1e-5f);
    float c0 = 0.f, c1 = 0.f;
    for (int c = 0; c < D_; ++c) {
      const float v = (xs[tid][c] - m) * inv * norm_g[c] + norm_b[c];
      xs[tid][c] = v;
      xb[tid][c] = (bf16)v;
      c0 += v * Wl[c * 2 + 0];
      c1 += v * Wl[c * 2 + 1];
    }
    co[tid][0] = c0 + bl[0];
    co[tid][1] = c1 + bl[1];
  }
  __syncthreads();

  const int wave = tid >> 5;
  const int l = lane_id();
  const int kh = l >> 4;
  const int ncol = l & 15;

  __builtin_prefetch(wqkvb, 0, 3);  // global_prefetch_b8
  __builtin_prefetch(w1b, 0, 3);

  bf16x16 afr[4];
#pragma unroll
  for (int kk = 0; kk < 4; ++kk) afr[kk] = load_a_bf16(&xb[0][kk * 32], D_ + 8);

  for (int ct = wave * 10; ct < wave * 10 + 10; ++ct) {
    f32x8 acc = zero8();
    const int col0 = ct * 16;
    if (col0 < 384) {  // ---- QKV ----
#pragma unroll
      for (int kk = 0; kk < 4; ++kk) {
        const bf16x16 bfr = load_b_bf16(wqkvb + (kk * 32) * 384 + col0, 384);
        acc = WMMA_BF16(afr[kk], bfr, acc);
      }
#pragma unroll
      for (int r = 0; r < 8; ++r) {
        const int row = r + 8 * kh;
        const long g = g0 + row;
        const int bt = (int)(g / N_);
        const int n  = (int)(g - (long)bt * N_);
        const int col = col0 + ncol;
        const float val = acc[r] + bqkv[col];
        bf16* dst = (col < 128) ? qw : ((col < 256) ? kw : vw);
        dst[((long)bt * NP + n) * F_ + (col & 127)] = (bf16)val;
      }
    } else {  // ---- h = relu(xn@W1 + b1 + coords@aff1_W + aff1_b) ----
      const int hc0 = col0 - 384;
#pragma unroll
      for (int kk = 0; kk < 4; ++kk) {
        const bf16x16 bfr = load_b_bf16(w1b + (kk * 32) * 256 + hc0, 256);
        acc = WMMA_BF16(afr[kk], bfr, acc);
      }
#pragma unroll
      for (int r = 0; r < 8; ++r) {
        const int row = r + 8 * kh;
        const int hc = hc0 + ncol;
        float val = acc[r] + b1[hc] + co[row][0] * aff1_W[hc] +
                    co[row][1] * aff1_W[256 + hc] + aff1_b[hc];
        val = fmaxf(val, 0.f);
        if (hc < 128) uw[(g0 + row) * F_ + hc] = val;
        else          vvs[row][hc - 128] = val;
      }
    }
  }
  __syncthreads();

  if (tid < 16) {
    float m = 0.f, s = 0.f;
    for (int c = 0; c < F_; ++c) { const float v = vvs[tid][c]; m += v; s += v * v; }
    m *= (1.f / F_);
    s = s * (1.f / F_) - m * m;
    const float inv = rsqrtf(s + 1e-5f);
    const long g = g0 + tid;
    const int bt = (int)(g / N_);
    const int n  = (int)(g - (long)bt * N_);
    bf16* dst = vnw + ((long)bt * NP + n) * F_;
    for (int c = 0; c < F_; ++c)
      dst[c] = (bf16)((vvs[tid][c] - m) * inv * sgu_g[c] + sgu_b[c]);
  }
}

// ---------------------------------------------------------------------------
// k2: tiny attention per (b,t): TDM-staged K/V panels in LDS, flash-style
//     online softmax, fused Wo projection.
// ---------------------------------------------------------------------------
__global__ __launch_bounds__(256) void k2_attn(
    const bf16* __restrict__ qw, const bf16* __restrict__ kw, const bf16* __restrict__ vw,
    const bf16* __restrict__ wob, const float* __restrict__ bo,
    float* __restrict__ gate) {
  __shared__ bf16 pbuf[8][16][32];
  __shared__ bf16 gbuf[8][16][F_ + 8];
  extern __shared__ bf16 panels[];             // [2 * NP * F_]: K panel, V panel
  bf16* kls = panels;
  bf16* vls = panels + NP * F_;

  const int bt = blockIdx.x;
  const int wave = (int)(threadIdx.x >> 5);
  const int l = lane_id();
  const int kh = l >> 4;
  const int ncol = l & 15;
  const bf16* qp = qw + (long)bt * NP * F_;
  const bf16* kp = kw + (long)bt * NP * F_;
  const bf16* vp = vw + (long)bt * NP * F_;

#if HAS_TDM
  if (wave == 0) {
    tdm_load_2d_bf16(kp, (u32)(unsigned long long)(uintptr_t)kls, F_, NP, F_);
    tdm_load_2d_bf16(vp, (u32)(unsigned long long)(uintptr_t)vls, F_, NP, F_);
    __builtin_amdgcn_s_wait_tensorcnt(0);
  }
#else
  for (int i = (int)threadIdx.x; i < NP * F_; i += 256) { kls[i] = kp[i]; vls[i] = vp[i]; }
#endif
  __syncthreads();

  for (int rt = wave; rt < NP / 16; rt += 8) {
    bf16x16 aq[4];
#pragma unroll
    for (int kk = 0; kk < 4; ++kk) aq[kk] = load_a_bf16(qp + rt * 16 * F_ + kk * 32, F_);

    f32x8 o[8];
#pragma unroll
    for (int j = 0; j < 8; ++j) o[j] = zero8();
    float rmax[8], rsum[8];
#pragma unroll
    for (int r = 0; r < 8; ++r) { rmax[r] = -3.0e38f; rsum[r] = 0.f; }

    for (int p = 0; p < NP / 32; ++p) {
      f32x8 s0 = zero8(), s1 = zero8();
#pragma unroll
      for (int half = 0; half < 2; ++half) {
        const int c = 2 * p + half;
#pragma unroll
        for (int kk = 0; kk < 4; ++kk) {
          // B = K^T tile from LDS: B[k][n] = K[c*16+n][kk*32+k], contiguous per lane
          const bf16* kr = kls + (c * 16 + ncol) * F_ + kk * 32 + kh * 16;
          bf16x16 bfr;
#pragma unroll
          for (int e = 0; e < 16; ++e) bfr[e] = kr[e];
          if (half == 0) s0 = WMMA_BF16(aq[kk], bfr, s0);
          else           s1 = WMMA_BF16(aq[kk], bfr, s1);
        }
      }
      const bool v0 = (2 * p) * 16 + ncol < N_;
      const bool v1 = (2 * p + 1) * 16 + ncol < N_;
#pragma unroll
      for (int r = 0; r < 8; ++r) {
        if (!v0) s0[r] = -1.0e30f;
        if (!v1) s1[r] = -1.0e30f;
      }
#pragma unroll
      for (int r = 0; r < 8; ++r) {
        float tm = fmaxf(s0[r], s1[r]);
#pragma unroll
        for (int msk = 1; msk < 16; msk <<= 1) tm = fmaxf(tm, __shfl_xor(tm, msk, 32));
        const float nm = fmaxf(rmax[r], tm);
        const float alpha = __expf(rmax[r] - nm);
        const float p0 = __expf(s0[r] - nm);
        const float p1 = __expf(s1[r] - nm);
        float ts = p0 + p1;
#pragma unroll
        for (int msk = 1; msk < 16; msk <<= 1) ts += __shfl_xor(ts, msk, 32);
        rsum[r] = rsum[r] * alpha + ts;
        rmax[r] = nm;
#pragma unroll
        for (int j = 0; j < 8; ++j) o[j][r] *= alpha;
        pbuf[wave][r + 8 * kh][ncol]      = (bf16)p0;
        pbuf[wave][r + 8 * kh][16 + ncol] = (bf16)p1;
      }
      const bf16x16 ap = load_a_bf16(&pbuf[wave][0][0], 32);
#pragma unroll
      for (int j = 0; j < 8; ++j) {
        const bf16x16 bv = load_b_bf16(vls + (p * 32) * F_ + j * 16, F_);
        o[j] = WMMA_BF16(ap, bv, o[j]);
      }
    }
#pragma unroll
    for (int r = 0; r < 8; ++r) {
      const float inv = 1.f / rsum[r];
#pragma unroll
      for (int j = 0; j < 8; ++j) gbuf[wave][r + 8 * kh][j * 16 + ncol] = (bf16)(o[j][r] * inv);
    }
    bf16x16 ag[4];
#pragma unroll
    for (int kk = 0; kk < 4; ++kk) ag[kk] = load_a_bf16(&gbuf[wave][0][kk * 32], F_ + 8);
#pragma unroll
    for (int j = 0; j < 8; ++j) {
      f32x8 acc = zero8();
#pragma unroll
      for (int kk = 0; kk < 4; ++kk) {
        const bf16x16 bw = load_b_bf16(wob + (kk * 32) * F_ + j * 16, F_);
        acc = WMMA_BF16(ag[kk], bw, acc);
      }
#pragma unroll
      for (int r = 0; r < 8; ++r) {
        const int row = rt * 16 + r + 8 * kh;
        if (row < N_) {
          const int col = j * 16 + ncol;
          gate[((long)bt * N_ + row) * F_ + col] = acc[r] + bo[col];
        }
      }
    }
  }
}

// ---------------------------------------------------------------------------
// k3: spatial-gate conv (3 dense NxN matmuls, vn panel TDM-staged in LDS)
//     + gate combine + FC2 + residual
// ---------------------------------------------------------------------------
__global__ __launch_bounds__(256) void k3_conv_ffn(
    const bf16* __restrict__ vnw, const bf16* __restrict__ wpad,
    const float* __restrict__ conv_b,
    const float* __restrict__ gate, const float* __restrict__ uw,
    const bf16* __restrict__ w2b, const float* __restrict__ b2,
    const float* __restrict__ x, float* __restrict__ out) {
  __shared__ bf16 sgbuf[8][16][F_ + 8];
  extern __shared__ bf16 vnls[];               // [NP * F_]

  const int bt = blockIdx.x;
  const int wave = (int)(threadIdx.x >> 5);
  const int l = lane_id();
  const int kh = l >> 4;
  const int ncol = l & 15;
  const bf16* vp = vnw + (long)bt * NP * F_;

  __builtin_prefetch(w2b, 0, 3);

#if HAS_TDM
  if (wave == 0) {
    tdm_load_2d_bf16(vp, (u32)(unsigned long long)(uintptr_t)vnls, F_, NP, F_);
    __builtin_amdgcn_s_wait_tensorcnt(0);
  }
#else
  for (int i = (int)threadIdx.x; i < NP * F_; i += 256) vnls[i] = vp[i];
#endif
  __syncthreads();

  for (int rt = wave; rt < NP / 16; rt += 8) {
    f32x8 acc[8];
#pragma unroll
    for (int j = 0; j < 8; ++j) acc[j] = zero8();

    for (int kt = 0; kt < NP / 32; ++kt) {
#pragma unroll
      for (int dk = 0; dk < 3; ++dk) {
        const bf16x16 aw = load_a_bf16(wpad + (long)dk * NP * NP + (rt * 16) * NP + kt * 32, NP);
#pragma unroll
        for (int j = 0; j < 8; ++j) {
          const int ff = j * 16 + ncol + dk - 1;
          bf16x16 bfr;
          if (ff >= 0 && ff < F_) {
            const bf16* src = vnls + (kt * 32 + kh * 16) * F_ + ff;
#pragma unroll
            for (int v = 0; v < 8; ++v) {
              bfr[2 * v]     = src[(2 * v) * F_];
              bfr[2 * v + 1] = src[(2 * v + 1) * F_];
            }
          } else {
#pragma unroll
            for (int e = 0; e < 16; ++e) bfr[e] = (bf16)0.0f;
          }
          acc[j] = WMMA_BF16(aw, bfr, acc[j]);
        }
      }
    }
#pragma unroll
    for (int j = 0; j < 8; ++j) {
#pragma unroll
      for (int r = 0; r < 8; ++r) {
        const int row = rt * 16 + r + 8 * kh;
        const int f = j * 16 + ncol;
        float val = 0.f;
        if (row < N_) {
          const long gi = ((long)bt * N_ + row) * F_ + f;
          val = (acc[j][r] + conv_b[row] + gate[gi]) * uw[gi];
        }
        sgbuf[wave][r + 8 * kh][f] = (bf16)val;
      }
    }
    bf16x16 as[4];
#pragma unroll
    for (int kk = 0; kk < 4; ++kk) as[kk] = load_a_bf16(&sgbuf[wave][0][kk * 32], F_ + 8);
#pragma unroll
    for (int j = 0; j < 8; ++j) {
      f32x8 a2 = zero8();
#pragma unroll
      for (int kk = 0; kk < 4; ++kk) {
        const bf16x16 bw = load_b_bf16(w2b + (kk * 32) * D_ + j * 16, D_);
        a2 = WMMA_BF16(as[kk], bw, a2);
      }
#pragma unroll
      for (int r = 0; r < 8; ++r) {
        const int row = rt * 16 + r + 8 * kh;
        if (row < N_) {
          const int col = j * 16 + ncol;
          const long gi = ((long)bt * N_ + row) * D_ + col;
          out[gi] = fmaxf(a2[r] + b2[col], 0.f) + x[gi];
        }
      }
    }
  }
}

// ---------------------------------------------------------------------------
extern "C" void kernel_launch(void* const* d_in, const int* in_sizes, int n_in,
                              void* d_out, int out_size, void* d_ws, size_t ws_size,
                              hipStream_t stream) {
  const float* x      = (const float*)d_in[0];
  const float* norm_g = (const float*)d_in[1];
  const float* norm_b = (const float*)d_in[2];
  const float* W1     = (const float*)d_in[3];
  const float* b1     = (const float*)d_in[4];
  const float* aff1_W = (const float*)d_in[5];
  const float* aff1_b = (const float*)d_in[6];
  const float* Wl     = (const float*)d_in[7];
  const float* bl     = (const float*)d_in[8];
  const float* sgu_g  = (const float*)d_in[9];
  const float* sgu_b  = (const float*)d_in[10];
  const float* conv_w = (const float*)d_in[11];
  const float* conv_b = (const float*)d_in[12];
  const float* Wqkv   = (const float*)d_in[13];
  const float* bqkv   = (const float*)d_in[14];
  const float* Wo     = (const float*)d_in[15];
  const float* bo     = (const float*)d_in[16];
  const float* W2     = (const float*)d_in[17];
  const float* b2     = (const float*)d_in[18];
  float* out = (float*)d_out;

  // workspace layout
  char* p = (char*)d_ws;
  const size_t SQ = (size_t)NBT * NP * F_ * sizeof(bf16);      // 34.6 MB each
  bf16* qw  = (bf16*)p; p += SQ;
  bf16* kw  = (bf16*)p; p += SQ;
  bf16* vw  = (bf16*)p; p += SQ;
  bf16* vnw = (bf16*)p; p += SQ;
  bf16* wpad  = (bf16*)p; p += ((size_t)3 * NP * NP * sizeof(bf16) + 255) & ~(size_t)255;
  bf16* wqkvb = (bf16*)p; p += (size_t)D_ * 384 * sizeof(bf16);
  bf16* w1b   = (bf16*)p; p += (size_t)D_ * 256 * sizeof(bf16);
  bf16* wob   = (bf16*)p; p += (size_t)D_ * 128 * sizeof(bf16);
  bf16* w2b   = (bf16*)p; p += (size_t)D_ * 128 * sizeof(bf16);
  p = (char*)(((uintptr_t)p + 255) & ~(uintptr_t)255);
  float* uw   = (float*)p; p += (size_t)NTOK * F_ * sizeof(float);
  float* gate = (float*)p; p += (size_t)NTOK * F_ * sizeof(float);

  {
    const long tot = 3L * NP * NP;
    k0_prep_w<<<(int)((tot + 255) / 256), 256, 0, stream>>>(conv_w, wpad);
  }
  k0_prep_dense<<<(D_ * 384 + 255) / 256, 256, 0, stream>>>(Wqkv, W1, Wo, W2,
                                                            wqkvb, w1b, wob, w2b);
  {
    const long tot = (long)NBT * (NP - N_) * F_;
    k0_zero_pad<<<(int)((tot + 255) / 256), 256, 0, stream>>>(qw, kw, vw, vnw);
  }
  k1_ln_proj<<<NTOK / 16, 128, 0, stream>>>(x, norm_g, norm_b, w1b, b1, aff1_W, aff1_b,
                                            Wl, bl, sgu_g, sgu_b, wqkvb, bqkv,
                                            qw, kw, vw, vnw, uw);
  k2_attn<<<NBT, 256, (size_t)2 * NP * F_ * sizeof(bf16), stream>>>(qw, kw, vw, wob, bo, gate);
  k3_conv_ffn<<<NBT, 256, (size_t)NP * F_ * sizeof(bf16), stream>>>(vnw, wpad, conv_b, gate,
                                                                    uw, w2b, b2, x, out);
}
